// TrajGRUCell_52372831207736
// MI455X (gfx1250) — matrix operand order repaired and checked by reading, not compiled, not run on hardware
//
#include <hip/hip_runtime.h>

// ---------------------------------------------------------------------------
// TrajGRU cell on gfx1250. All convs = implicit GEMM via
// v_wmma_f32_16x16x32_f16. Activations kept in channel-tiled f16 layout
// [b][C/16][H][W][16] so every WMMA B-fragment is 2 contiguous b128 loads.
// Weights packed K-major f16 so A-fragments are 2 contiguous b128 loads.
// GEMM waves process 2 M-tiles per B-fragment (dual accumulators) to halve
// B traffic. Deformable conv: coordinate-shared gather (16 channels/thread)
// into a fragment-ordered f16 val buffer, then a streaming WMMA GEMM.
// ---------------------------------------------------------------------------

typedef __attribute__((ext_vector_type(16))) _Float16 v16h;
typedef __attribute__((ext_vector_type(8)))  _Float16 v8h;
typedef __attribute__((ext_vector_type(8)))  float    v8f;

constexpr int H_ = 128;
constexpr int W_ = 128;
constexpr int B_ = 4;
constexpr size_t HW16 = (size_t)H_ * W_ * 16;  // halves per channel-block

// ---- A fragment: lane l holds row m = l&15; element i -> K = i+16*hi.
__device__ __forceinline__ v16h load_a16(const _Float16* __restrict__ p) {
  v16h a;
#pragma unroll
  for (int i = 0; i < 16; ++i) a[i] = p[i];
  return a;
}

// ---- B fragment from channel-tiled f16 tensors (optionally concatenated).
// Per (cb,hi) a lane covers one 16-channel block: blk = cb*2 + hi. Blocks
// [0,NB1) come from T1b, [NB1,NB1+NB2) from T2b, rest are zero padding.
// T1b/T2b are already offset to batch b. 16 contiguous halves = 2 b128 loads.
template <int NB1, int NB2, int KSZ, int PAD>
__device__ __forceinline__ v16h build_b3(const _Float16* __restrict__ T1b,
                                         const _Float16* __restrict__ T2b,
                                         int y, int xg, int hi,
                                         int kh, int kw, int cb) {
  const int ya = y + kh - PAD;
  const int xa = xg + kw - PAD;
  const bool ok = (ya >= 0) && (ya < H_) && (xa >= 0) && (xa < W_);
  const int blk = cb * 2 + hi;
  const _Float16* src = nullptr;
  if (blk < NB1)            src = T1b + ((size_t)blk * H_ * W_ + (size_t)ya * W_ + xa) * 16;
  else if (blk < NB1 + NB2) src = T2b + ((size_t)(blk - NB1) * H_ * W_ + (size_t)ya * W_ + xa) * 16;
  v16h bf;
  if (src != nullptr && ok) {
#pragma unroll
    for (int i = 0; i < 16; ++i) bf[i] = src[i];
  } else {
#pragma unroll
    for (int i = 0; i < 16; ++i) bf[i] = (_Float16)0.f;
  }
  return bf;
}

// One wave = one pixel tile x MTW m-tile-group; gw is wave-uniform so EXEC
// stays all-ones inside active waves (WMMA requirement).
#define TILE_DECOMP(MTW_)                                                     \
  const int lane = threadIdx.x & 31;                                          \
  const int gw   = (int)((blockIdx.x * blockDim.x + threadIdx.x) >> 5);       \
  if (gw >= B_ * ((H_ * W_) / 16) * (MTW_)) return;                           \
  const int mtw = gw % (MTW_);                                                \
  const int pt  = gw / (MTW_);                                                \
  const int b   = pt >> 10;                                                   \
  const int ptl = pt & 1023;                                                  \
  const int y   = ptl >> 3;                                                   \
  const int x0  = (ptl & 7) << 4;                                             \
  const int ln  = lane & 15, hi = lane >> 4;                                  \
  const int xg  = x0 + ln;

// --------------------------- packing kernels -------------------------------
// K-major weight pack: dst[m][kk*CINP + c] = w[m][c][kh][kw], zero-padded.
__global__ void pack_w_kmajor_kernel(const float* __restrict__ src,
                                     _Float16* __restrict__ dst,
                                     int cout, int CIN, int KK, int CINP, int n) {
  int idx = blockIdx.x * blockDim.x + threadIdx.x;
  if (idx >= n) return;
  int m   = idx / (KK * CINP);
  int rem = idx - m * (KK * CINP);
  int kk  = rem / CINP;
  int c   = rem - kk * CINP;
  float v = (m < cout && c < CIN) ? src[((size_t)m * CIN + c) * KK + kk] : 0.f;
  dst[idx] = (_Float16)v;
}

// Pack concat(input16, state64) into tiled f16 [b][5][H][W][16].
__global__ void pack_instate_kernel(const float* __restrict__ input,
                                    const float* __restrict__ state,
                                    _Float16* __restrict__ dst, int n) {
  int idx = blockIdx.x * blockDim.x + threadIdx.x;
  if (idx >= n) return;
  int ci   = idx & 15;
  int rest = idx >> 4;
  int x    = rest & (W_ - 1);  rest >>= 7;
  int y    = rest & (H_ - 1);  rest >>= 7;
  int blk  = rest % 5;
  int b    = rest / 5;
  float v = (blk == 0)
      ? input[(((size_t)b * 16 + ci) * H_ + y) * W_ + x]
      : state[(((size_t)b * 64 + (blk - 1) * 16 + ci) * H_ + y) * W_ + x];
  dst[idx] = (_Float16)v;
}

// --------------------------- generic conv ----------------------------------
// OUTMODE 0: plain f32 NCHW. OUTMODE 1: channel-tiled f16 (one b128/lane).
// MTPW m-tiles per wave share each B-fragment (dual accumulators).
template <int NB1, int NB2, int COUT, int KSZ, int PAD,
          int ACT /*0=none,1=leaky0.2*/, int OUTMODE, int MTPW>
__global__ void conv_wmma_kernel(const _Float16* __restrict__ T1, int bstr1,
                                 const _Float16* __restrict__ T2, int bstr2,
                                 const _Float16* __restrict__ Wpk,
                                 const float* __restrict__ bias,
                                 void* __restrict__ outp) {
  constexpr int KK   = KSZ * KSZ;
  constexpr int NCB  = (NB1 + NB2 + 1) / 2;   // 32-channel chunks (padded)
  constexpr int CINP = NCB * 32;
  constexpr int R32  = KK * CINP;
  constexpr int MT   = (COUT + 15) / 16;
  static_assert(MT % MTPW == 0, "MT must be divisible by MTPW");
  constexpr int MTW  = MT / MTPW;
  TILE_DECOMP(MTW)
  const int mt0 = mtw * MTPW;
  const _Float16* T1b = T1 + (size_t)b * bstr1 * HW16;
  const _Float16* T2b = (NB2 > 0) ? (T2 + (size_t)b * bstr2 * HW16) : nullptr;
  const _Float16* wrow[MTPW];
#pragma unroll
  for (int t = 0; t < MTPW; ++t)
    wrow[t] = Wpk + (size_t)((mt0 + t) * 16 + ln) * R32 + 16 * hi;
  const v8f vzero = {};
  v8f acc[MTPW];
#pragma unroll
  for (int t = 0; t < MTPW; ++t) acc[t] = vzero;
#pragma unroll
  for (int kh = 0; kh < KSZ; ++kh) {
#pragma unroll
    for (int kw = 0; kw < KSZ; ++kw) {
#pragma unroll
      for (int cb = 0; cb < NCB; ++cb) {
        v16h bf = build_b3<NB1, NB2, KSZ, PAD>(T1b, T2b, y, xg, hi, kh, kw, cb);
#pragma unroll
        for (int t = 0; t < MTPW; ++t) {
          v16h a = load_a16(wrow[t] + (kh * KSZ + kw) * CINP + cb * 32);
          acc[t] = __builtin_amdgcn_wmma_f32_16x16x32_f16(false, a, false, bf,
                                                          (short)0, acc[t],
                                                          false, false);
        }
      }
    }
  }
#pragma unroll
  for (int t = 0; t < MTPW; ++t) {
    const int mt = mt0 + t;
    if (OUTMODE == 1) {
      // tiled f16 store: lane writes 8 contiguous halves (16B) at +8*hi
      v8h sv;
#pragma unroll
      for (int j = 0; j < 8; ++j) {
        int m = mt * 16 + j + 8 * hi;
        float v = acc[t][j] + bias[m];
        if (ACT == 1) v = v > 0.f ? v : 0.2f * v;
        sv[j] = (_Float16)v;
      }
      size_t base = (((size_t)b * (COUT / 16) + mt) * H_ * W_ +
                     (size_t)y * W_ + (x0 + ln)) * 16 + 8 * hi;
      *(v8h*)((_Float16*)outp + base) = sv;
    } else {
#pragma unroll
      for (int j = 0; j < 8; ++j) {
        int m = mt * 16 + j + 8 * hi;
        if (COUT % 16 == 0 || m < COUT) {
          float v = acc[t][j] + bias[m];
          if (ACT == 1) v = v > 0.f ? v : 0.2f * v;
          size_t oi = (((size_t)b * COUT + m) * H_ + y) * W_ + (x0 + ln);
          ((float*)outp)[oi] = v;
        }
      }
    }
  }
}

// --------------------------- deformable conv -------------------------------
// w_def packed K-major (r = kk*64 + c), so K-chunk kc covers kk = kc>>1 and
// channels c = (kc&1)*32 + 16*hi + i  (16 consecutive channels per lane-half).
// One gather thread = one (tile, kc, lane): computes bilinear coords ONCE,
// reads 4 corners x 16 contiguous f16 channels from tiled flows, writes 16
// contiguous halves of valpk in exact B-fragment order.
__global__ void deform_gather_kernel(const _Float16* __restrict__ FLWT,  // [b][4][H][W][16]
                                     const float* __restrict__ OFF,      // [b][18][H][W] f32
                                     _Float16* __restrict__ valpk, int total) {
  int idx = blockIdx.x * blockDim.x + threadIdx.x;
  if (idx >= total) return;
  int lane = idx & 31;
  int tc   = idx >> 5;          // tile*18 + kc
  int kc   = tc % 18;
  int pt   = tc / 18;
  int b    = pt >> 10;
  int ptl  = pt & 1023;
  int y    = ptl >> 3;
  int x0   = (ptl & 7) << 4;
  int hi   = lane >> 4;
  int xg   = x0 + (lane & 15);
  int kk   = kc >> 1;
  int blk  = ((kc & 1) << 1) + hi;      // 16-channel block 0..3
  float offr = OFF[(((size_t)b * 18 + kk) * H_ + y) * W_ + xg];
  float offc = OFF[(((size_t)b * 18 + 9 + kk) * H_ + y) * W_ + xg];
  float pr = (float)(y + 1)  + (float)(kk / 3 - 1) + offr;
  float pc = (float)(xg + 1) + (float)(kk % 3 - 1) + offc;
  pr = fminf(fmaxf(pr, 0.f), 129.f);
  pc = fminf(fmaxf(pc, 0.f), 129.f);
  float r0f = floorf(pr), c0f = floorf(pc);
  float r1f = fminf(r0f + 1.f, 129.f), c1f = fminf(c0f + 1.f, 129.f);
  int r0 = (int)r0f, c0 = (int)c0f, r1 = (int)r1f, c1 = (int)c1f;
  float wr0 = 1.f - (pr - r0f);
  float wc0 = 1.f - (pc - c0f);
  float wr1 = 1.f - (r1f - pr);
  float wc1 = 1.f - (c1f - pc);
  const _Float16* base = FLWT + ((size_t)b * 4 + blk) * HW16;
  auto corner = [&](int rp, int cp) -> const _Float16* {
    int ri = rp - 1, ci = cp - 1;    // padded [0,129] -> unpadded
    if (ri < 0 || ri >= H_ || ci < 0 || ci >= W_) return nullptr;
    return base + ((size_t)ri * W_ + ci) * 16;
  };
  const _Float16* p00 = corner(r0, c0);
  const _Float16* p01 = corner(r0, c1);
  const _Float16* p10 = corner(r1, c0);
  const _Float16* p11 = corner(r1, c1);
  _Float16* dst = valpk + (size_t)idx * 16;
#pragma unroll
  for (int i = 0; i < 16; ++i) {
    float v = 0.f;
    if (p00) v += wr0 * wc0 * (float)p00[i];
    if (p01) v += wr0 * wc1 * (float)p01[i];
    if (p10) v += wr1 * wc0 * (float)p10[i];
    if (p11) v += wr1 * wc1 * (float)p11[i];
    dst[i] = (_Float16)v;
  }
}

// Phase 2: pure GEMM; B-fragments are contiguous b128 loads from valpk,
// shared across 2 M-tiles (dual accumulators). Output tiled f16.
__global__ void deform_gemm_kernel(const _Float16* __restrict__ valpk,
                                   const _Float16* __restrict__ Wpk,   // 64 x 576, k-major
                                   _Float16* __restrict__ outp) {
  constexpr int MTW = 2, NC = 18, R32 = 576;
  TILE_DECOMP(MTW)
  const int mt0 = mtw * 2;
  const _Float16* bp = valpk + (((size_t)pt * NC) * 32 + lane) * 16;
  const _Float16* w0 = Wpk + (size_t)(mt0 * 16 + ln) * R32 + 16 * hi;
  const _Float16* w1 = Wpk + (size_t)((mt0 + 1) * 16 + ln) * R32 + 16 * hi;
  v8f acc0 = {}, acc1 = {};
#pragma unroll
  for (int kc = 0; kc < NC; ++kc) {
    v16h bf;
#pragma unroll
    for (int i = 0; i < 16; ++i) bf[i] = bp[(size_t)kc * 512 + i];
    v16h a0 = load_a16(w0 + kc * 32);
    v16h a1 = load_a16(w1 + kc * 32);
    acc0 = __builtin_amdgcn_wmma_f32_16x16x32_f16(false, a0, false, bf,
                                                  (short)0, acc0, false, false);
    acc1 = __builtin_amdgcn_wmma_f32_16x16x32_f16(false, a1, false, bf,
                                                  (short)0, acc1, false, false);
  }
#pragma unroll
  for (int t = 0; t < 2; ++t) {
    const v8f& acc = t ? acc1 : acc0;
    v8h sv;
#pragma unroll
    for (int j = 0; j < 8; ++j) sv[j] = (_Float16)acc[j];
    size_t base = (((size_t)b * 4 + (mt0 + t)) * H_ * W_ +
                   (size_t)y * W_ + (x0 + ln)) * 16 + 8 * hi;
    *(v8h*)(outp + base) = sv;
  }
}

// --------------------------- fused tail ------------------------------------
// z & r share one B-fragment per K-step (two A rows of w_conv); plus w_ri and
// w_rh GEMMs; plus sigmoid / leaky-ReLU gating -> final output (f32 NCHW).
__global__ void final_fused_kernel(const _Float16* __restrict__ INST,  // [b][5][H][W][16] (blk0 = input)
                                   const _Float16* __restrict__ WRPT,  // [b][4][H][W][16]
                                   const float* __restrict__ STATE,
                                   const _Float16* __restrict__ Wconv, // 128 x 864
                                   const _Float16* __restrict__ Wri,   // 64 x 288
                                   const _Float16* __restrict__ Wrh,   // 64 x 576
                                   const float* __restrict__ bconv,
                                   const float* __restrict__ bri,
                                   const float* __restrict__ brh,
                                   float* __restrict__ outp) {
  constexpr int MTW = 4;  // 64 output channels, 1 m-tile/wave (4 accumulators)
  TILE_DECOMP(MTW)
  const int mt = mtw;
  const int arow = mt * 16 + ln;
  const _Float16* INb = INST + (size_t)b * 5 * HW16;  // block 0 = input
  const _Float16* WRb = WRPT + (size_t)b * 4 * HW16;
  const _Float16* wz = Wconv + (size_t)arow * 864 + 16 * hi;
  const _Float16* wr = Wconv + (size_t)(64 + arow) * 864 + 16 * hi;
  const _Float16* wi = Wri + (size_t)arow * 288 + 16 * hi;
  const _Float16* wh = Wrh + (size_t)arow * 576 + 16 * hi;
  v8f accz = {}, accr = {}, accri = {}, accrh = {};
  // z and r: concat(input, wrapped) = blocks [input, wrapped*4, pad], 27 chunks.
#pragma unroll
  for (int kh = 0; kh < 3; ++kh)
#pragma unroll
    for (int kw = 0; kw < 3; ++kw)
#pragma unroll
      for (int cb = 0; cb < 3; ++cb) {
        v16h bf = build_b3<1, 4, 3, 1>(INb, WRb, y, xg, hi, kh, kw, cb);
        v16h az = load_a16(wz + (kh * 3 + kw) * 96 + cb * 32);
        v16h ar = load_a16(wr + (kh * 3 + kw) * 96 + cb * 32);
        accz = __builtin_amdgcn_wmma_f32_16x16x32_f16(false, az, false, bf,
                                                      (short)0, accz, false, false);
        accr = __builtin_amdgcn_wmma_f32_16x16x32_f16(false, ar, false, bf,
                                                      (short)0, accr, false, false);
      }
  // ri: input only (block 0 of INST), 9 chunks.
#pragma unroll
  for (int kh = 0; kh < 3; ++kh)
#pragma unroll
    for (int kw = 0; kw < 3; ++kw) {
      v16h bf = build_b3<1, 0, 3, 1>(INb, (const _Float16*)nullptr,
                                     y, xg, hi, kh, kw, 0);
      v16h a = load_a16(wi + (kh * 3 + kw) * 32);
      accri = __builtin_amdgcn_wmma_f32_16x16x32_f16(false, a, false, bf,
                                                     (short)0, accri, false, false);
    }
  // rh: wrapped only, 18 chunks.
#pragma unroll
  for (int kh = 0; kh < 3; ++kh)
#pragma unroll
    for (int kw = 0; kw < 3; ++kw)
#pragma unroll
      for (int cb = 0; cb < 2; ++cb) {
        v16h bf = build_b3<4, 0, 3, 1>(WRb, (const _Float16*)nullptr,
                                       y, xg, hi, kh, kw, cb);
        v16h a = load_a16(wh + (kh * 3 + kw) * 64 + cb * 32);
        accrh = __builtin_amdgcn_wmma_f32_16x16x32_f16(false, a, false, bf,
                                                       (short)0, accrh, false, false);
      }
#pragma unroll
  for (int j = 0; j < 8; ++j) {
    int m = mt * 16 + j + 8 * hi;
    float z  = 1.f / (1.f + __expf(-(accz[j] + bconv[m])));
    float rr = 1.f / (1.f + __expf(-(accr[j] + bconv[64 + m])));
    float hp = (accri[j] + bri[m]) + rr * (accrh[j] + brh[m]);
    float h_ = hp > 0.f ? hp : 0.2f * hp;
    size_t oi = (((size_t)b * 64 + m) * H_ + y) * W_ + (x0 + ln);
    float h = STATE[oi];
    outp[oi] = (1.f - z) * h_ + z * h;
  }
}

// --------------------------- launcher --------------------------------------
extern "C" void kernel_launch(void* const* d_in, const int* in_sizes, int n_in,
                              void* d_out, int out_size, void* d_ws, size_t ws_size,
                              hipStream_t stream) {
  (void)in_sizes; (void)n_in; (void)out_size; (void)ws_size;
  const float* input  = (const float*)d_in[0];
  const float* state  = (const float*)d_in[1];
  const float* w_uv1  = (const float*)d_in[2];
  const float* b_uv1  = (const float*)d_in[3];
  const float* w_uv2  = (const float*)d_in[4];
  const float* b_uv2  = (const float*)d_in[5];
  const float* w_off  = (const float*)d_in[6];
  const float* b_off  = (const float*)d_in[7];
  const float* w_def  = (const float*)d_in[8];
  const float* w_conv = (const float*)d_in[9];
  const float* b_conv = (const float*)d_in[10];
  const float* w_ri   = (const float*)d_in[11];
  const float* b_ri   = (const float*)d_in[12];
  const float* w_rh   = (const float*)d_in[13];
  const float* b_rh   = (const float*)d_in[14];
  float* out = (float*)d_out;

  char* ws = (char*)d_ws;
  size_t off = 0;
  auto carve = [&](size_t bytes) {
    char* p = ws + off;
    off = (off + bytes + 255) & ~(size_t)255;
    return p;
  };
  _Float16* pk_uv1 = (_Float16*)carve((size_t)32  * 25 * 96 * 2);  // 32 x 2400
  _Float16* pk_uv2 = (_Float16*)carve((size_t)64  * 25 * 32 * 2);  // 64 x 800
  _Float16* pk_off = (_Float16*)carve((size_t)32  * 9  * 64 * 2);  // 32 x 576
  _Float16* pk_def = (_Float16*)carve((size_t)64  * 9  * 64 * 2);  // 64 x 576 k-major
  _Float16* pk_cnv = (_Float16*)carve((size_t)128 * 9  * 96 * 2);  // 128 x 864
  _Float16* pk_ri  = (_Float16*)carve((size_t)64  * 9  * 32 * 2);  // 64 x 288
  _Float16* pk_rh  = (_Float16*)carve((size_t)64  * 9  * 64 * 2);  // 64 x 576
  _Float16* instate = (_Float16*)carve((size_t)B_ * 5 * HW16 * 2); // tiled concat
  _Float16* flows1t = (_Float16*)carve((size_t)B_ * 2 * HW16 * 2);
  _Float16* flowst  = (_Float16*)carve((size_t)B_ * 4 * HW16 * 2);
  float*    offs    = (float*)   carve((size_t)B_ * 18 * H_ * W_ * 4);
  _Float16* wrappedt= (_Float16*)carve((size_t)B_ * 4 * HW16 * 2);
  _Float16* valpk   = (_Float16*)carve((size_t)B_ * 1024 * 18 * 512 * 2); // 75.5 MB

  auto packk = [&](const float* src, _Float16* dst, int cout, int cin, int KK,
                   int cinp, int cout16) {
    int n = cout16 * KK * cinp;
    pack_w_kmajor_kernel<<<dim3((n + 255) / 256), dim3(256), 0, stream>>>(
        src, dst, cout, cin, KK, cinp, n);
  };
  packk(w_uv1,  pk_uv1, 32,  80, 25, 96, 32);
  packk(w_uv2,  pk_uv2, 64,  32, 25, 32, 64);
  packk(w_off,  pk_off, 18,  64, 9,  64, 32);
  packk(w_def,  pk_def, 64,  64, 9,  64, 64);
  packk(w_conv, pk_cnv, 128, 80, 9,  96, 128);
  packk(w_ri,   pk_ri,  64,  16, 9,  32, 64);
  packk(w_rh,   pk_rh,  64,  64, 9,  64, 64);
  {
    int n = B_ * 5 * (int)HW16;
    pack_instate_kernel<<<dim3((n + 255) / 256), dim3(256), 0, stream>>>(
        input, state, instate, n);
  }

  auto grid_for = [](int MTW) {
    int waves = B_ * ((H_ * W_) / 16) * MTW;      // one tile-group per wave
    return dim3((unsigned)(waves * 32 / 256));    // 8 waves / 256-thread block
  };

  // conv1: concat(input,state) 80ch -> 32ch, 5x5, leaky 0.2, tiled f16 out
  conv_wmma_kernel<5, 0, 32, 5, 2, 1, 1, 2>
      <<<grid_for(1), 256, 0, stream>>>(instate, 5, (const _Float16*)nullptr, 0,
                                        pk_uv1, b_uv1, (void*)flows1t);
  // conv2: flows1 32ch -> 64ch, 5x5, tiled f16 out
  conv_wmma_kernel<2, 0, 64, 5, 2, 0, 1, 2>
      <<<grid_for(2), 256, 0, stream>>>(flows1t, 2, (const _Float16*)nullptr, 0,
                                        pk_uv2, b_uv2, (void*)flowst);
  // offsets: flows 64ch -> 18ch, 3x3, plain f32 out
  conv_wmma_kernel<4, 0, 18, 3, 1, 0, 0, 2>
      <<<grid_for(1), 256, 0, stream>>>(flowst, 4, (const _Float16*)nullptr, 0,
                                        pk_off, b_off, (void*)offs);
  // deformable conv: coord-shared gather -> packed f16 val, then WMMA GEMM
  {
    int total = B_ * 1024 * 18 * 32;   // one thread per (tile, kc, lane)
    deform_gather_kernel<<<dim3((total + 255) / 256), dim3(256), 0, stream>>>(
        flowst, offs, valpk, total);
  }
  deform_gemm_kernel<<<grid_for(2), 256, 0, stream>>>(valpk, pk_def, wrappedt);
  // fused tail: z/r/ri/rh GEMMs + gating -> output
  final_fused_kernel<<<grid_for(4), 256, 0, stream>>>(
      instate, wrappedt, state, pk_cnv, pk_ri, pk_rh, b_conv, b_ri, b_rh, out);
}
